// Qwen3Model_19456201851390
// MI455X (gfx1250) — matrix operand order repaired and checked by reading, not compiled, not run on hardware
//
#include <hip/hip_runtime.h>
#include <cstdint>
#include <cstddef>

typedef __bf16 bf16;
typedef __attribute__((ext_vector_type(16))) __bf16 v16bf;
typedef __attribute__((ext_vector_type(8)))  __bf16 v8bf;
typedef __attribute__((ext_vector_type(8)))  float  v8f;

namespace {
constexpr int Vn  = 32000;
constexpr int Dm  = 1024;
constexpr int Hn  = 16;
constexpr int KVn = 4;
constexpr int HDn = 64;
constexpr int Ln  = 8;
constexpr int FFn = 4096;
constexpr int Bn  = 2;
constexpr int Sn  = 1024;
constexpr int Tn  = Bn * Sn;     // 2048 tokens
constexpr int GRPn = Hn / KVn;   // 4
}

// ---------------------------------------------------------------------------
// Embedding gather + sqrt(D) scale.  x[t, :] = emb[ids[t], :] * 32
// ---------------------------------------------------------------------------
__global__ __launch_bounds__(256) void embed_kernel(const int* __restrict__ ids,
                                                    const float* __restrict__ emb,
                                                    float* __restrict__ x) {
    const int t = blockIdx.x, tid = threadIdx.x;
    const int id = ids[t];
    float4 e = *(const float4*)(emb + (size_t)id * Dm + tid * 4);
    float4 o; o.x = e.x * 32.f; o.y = e.y * 32.f; o.z = e.z * 32.f; o.w = e.w * 32.f;
    *(float4*)(x + (size_t)t * Dm + tid * 4) = o;
}

// ---------------------------------------------------------------------------
// Token RMSNorm over D=1024, output bf16 (GEMM A-operand).  1 block / token.
// ---------------------------------------------------------------------------
__global__ __launch_bounds__(256) void rmsnorm_bf16_kernel(const float* __restrict__ x,
                                                           const float* __restrict__ w,
                                                           bf16* __restrict__ out) {
    const int t = blockIdx.x, tid = threadIdx.x;
    const float* px = x + (size_t)t * Dm;
    float4 xv = *(const float4*)(px + tid * 4);
    __shared__ float red[256];
    red[tid] = xv.x * xv.x + xv.y * xv.y + xv.z * xv.z + xv.w * xv.w;
    __syncthreads();
    for (int off = 128; off > 0; off >>= 1) {
        if (tid < off) red[tid] += red[tid + off];
        __syncthreads();
    }
    const float rn = rsqrtf(red[0] * (1.0f / Dm) + 1e-6f);
    float4 wv = *(const float4*)(w + tid * 4);
    bf16* po = out + (size_t)t * Dm + tid * 4;
    po[0] = (bf16)(xv.x * rn * wv.x);
    po[1] = (bf16)(xv.y * rn * wv.y);
    po[2] = (bf16)(xv.z * rn * wv.z);
    po[3] = (bf16)(xv.w * rn * wv.w);
}

// ---------------------------------------------------------------------------
// Per-head RMSNorm (HD=64) + RoPE.  1 block (64 threads) per (token, head).
// in/out layout: [T, NH, HD] fp32.  pos = token % S.
// ---------------------------------------------------------------------------
__global__ __launch_bounds__(64) void rmsnorm_rope_kernel(const float* __restrict__ in,
                                                          const float* __restrict__ w,
                                                          float* __restrict__ out,
                                                          int NH) {
    const int idx = blockIdx.x;          // t*NH + h
    const int d = threadIdx.x;           // 0..63
    const int pos = (idx / NH) % Sn;
    const float* px = in + (size_t)idx * HDn;
    __shared__ float sh[64];
    __shared__ float xn[64];
    const float xv = px[d];
    sh[d] = xv * xv;
    __syncthreads();
    for (int off = 32; off > 0; off >>= 1) {
        if (d < off) sh[d] += sh[d + off];
        __syncthreads();
    }
    const float rn = rsqrtf(sh[0] * (1.0f / HDn) + 1e-6f);
    const float v = xv * rn * w[d];
    xn[d] = v;
    __syncthreads();
    const int dm = d & 31;
    const float inv_freq = __powf(10000.0f, -(float)(2 * dm) * (1.0f / HDn));
    const float ang = (float)pos * inv_freq;
    float s, c;
    __sincosf(ang, &s, &c);
    const float rot = (d < 32) ? -xn[d + 32] : xn[d - 32];
    out[(size_t)idx * HDn + d] = v * c + rot * s;
}

// ---------------------------------------------------------------------------
// Causal GQA attention, fp32.  1 block (128 thr) per (b, h, qpos).
// q: [B,S,H,HD]  k,v: [B,S,KV,HD]  ctx: [B,S,H,HD]
// ---------------------------------------------------------------------------
__global__ __launch_bounds__(128) void attention_kernel(const float* __restrict__ q,
                                                        const float* __restrict__ k,
                                                        const float* __restrict__ v,
                                                        float* __restrict__ ctx) {
    const int qpos = blockIdx.x, h = blockIdx.y, b = blockIdx.z;
    const int tid = threadIdx.x;
    const int kvh = h / GRPn;
    __shared__ float qs[HDn];
    __shared__ float sc[Sn];
    __shared__ float red[128];
    if (tid < HDn)
        qs[tid] = q[(((size_t)b * Sn + qpos) * Hn + h) * HDn + tid] * 0.125f;  // HD^-0.5
    __syncthreads();
    const int nk = qpos + 1;
    float lmax = -3.4e38f;
    for (int kk = tid; kk < nk; kk += 128) {
        const float* kp = k + (((size_t)b * Sn + kk) * KVn + kvh) * HDn;
        float dot = 0.f;
#pragma unroll
        for (int d0 = 0; d0 < HDn; d0 += 4) {
            float4 kv = *(const float4*)(kp + d0);
            dot += qs[d0] * kv.x + qs[d0 + 1] * kv.y + qs[d0 + 2] * kv.z + qs[d0 + 3] * kv.w;
        }
        sc[kk] = dot;
        lmax = fmaxf(lmax, dot);
    }
    red[tid] = lmax;
    __syncthreads();
    for (int off = 64; off > 0; off >>= 1) {
        if (tid < off) red[tid] = fmaxf(red[tid], red[tid + off]);
        __syncthreads();
    }
    const float m = red[0];
    __syncthreads();
    float lsum = 0.f;
    for (int kk = tid; kk < nk; kk += 128) {
        const float p = __expf(sc[kk] - m);
        sc[kk] = p;
        lsum += p;
    }
    red[tid] = lsum;
    __syncthreads();
    for (int off = 64; off > 0; off >>= 1) {
        if (tid < off) red[tid] += red[tid + off];
        __syncthreads();
    }
    const float inv = 1.0f / red[0];
    __syncthreads();
    const int d = tid & 63, slice = tid >> 6;
    float acc = 0.f;
    for (int kk = slice; kk < nk; kk += 2)
        acc += sc[kk] * v[(((size_t)b * Sn + kk) * KVn + kvh) * HDn + d];
    red[tid] = acc;
    __syncthreads();
    if (tid < 64)
        ctx[(((size_t)b * Sn + qpos) * Hn + h) * HDn + d] = (red[tid] + red[tid + 64]) * inv;
}

// ---------------------------------------------------------------------------
// Tiled transpose + fp32 -> bf16 convert.  W: K x N row-major  ->  Wt: N x K.
// block (32, 8), grid (N/32, K/32).
// ---------------------------------------------------------------------------
__global__ __launch_bounds__(256) void transpose_cvt_kernel(const float* __restrict__ W,
                                                            bf16* __restrict__ Wt,
                                                            int K, int N) {
    __shared__ float tile[32][33];
    const int nBase = blockIdx.x * 32;
    const int kBase = blockIdx.y * 32;
    const int tx = threadIdx.x;   // 0..31
    const int ty = threadIdx.y;   // 0..7
#pragma unroll
    for (int i = 0; i < 32; i += 8)
        tile[ty + i][tx] = W[(size_t)(kBase + ty + i) * N + nBase + tx];
    __syncthreads();
#pragma unroll
    for (int i = 0; i < 32; i += 8)
        Wt[(size_t)(nBase + ty + i) * K + kBase + tx] = (bf16)tile[tx][ty + i];
}

// ---------------------------------------------------------------------------
// WMMA fragment loaders (layouts per ISA 7.12.2):
//   A lane<16: k 0..7 & 16..23 ; lane>=16: k 8..15 & 24..31   (row = lane%16)
//   B lane<16: k 0..7 & 16..23 ; lane>=16: k 8..15 & 24..31   (col = lane%16)
// ---------------------------------------------------------------------------
__device__ __forceinline__ v16bf load_a_frag(const bf16* __restrict__ row, int k0, int hf) {
    v8bf lo = *(const v8bf*)(row + k0 + hf * 8);
    v8bf hi = *(const v8bf*)(row + k0 + 16 + hf * 8);
    return __builtin_shufflevector(lo, hi,
        0, 1, 2, 3, 4, 5, 6, 7, 8, 9, 10, 11, 12, 13, 14, 15);
}

__device__ __forceinline__ v16bf load_b_frag(const bf16* __restrict__ row, int k0, int hf) {
    v8bf lo = *(const v8bf*)(row + k0 + hf * 8);
    v8bf hi = *(const v8bf*)(row + k0 + 16 + hf * 8);
    return __builtin_shufflevector(lo, hi,
        0, 1, 2, 3, 4, 5, 6, 7, 8, 9, 10, 11, 12, 13, 14, 15);
}

// ---------------------------------------------------------------------------
// WMMA GEMM:  C[M,N] (f32) = A[M,K] (bf16, row-major) x Bt[N,K]^T (bf16).
// Each wave: one 32x64 output tile = 2 A-frags x 4 B-frags -> 8 accumulators.
// Per 32-K step: 12 b128 loads for 8 WMMAs (1.5 loads/WMMA).
// Block: 256 threads = 8 waves arranged 2(M) x 4(N) -> 64 x 256 block tile.
// Grid (N/256, M/64).
// ---------------------------------------------------------------------------
__global__ __launch_bounds__(256) void gemm_bf16_wmma(const bf16* __restrict__ A,
                                                      const bf16* __restrict__ Bt,
                                                      float* __restrict__ C,
                                                      int M, int N, int K) {
    const int lane = threadIdx.x & 31;
    const int wave = threadIdx.x >> 5;   // 0..7
    const int wm = wave >> 2;            // 0..1
    const int wn = wave & 3;             // 0..3
    const int m0 = blockIdx.y * 64 + wm * 32;
    const int n0 = blockIdx.x * 256 + wn * 64;
    const int r = lane & 15;
    const int hf = lane >> 4;

    const bf16* arow0 = A + (size_t)(m0 + r) * K;
    const bf16* arow1 = A + (size_t)(m0 + 16 + r) * K;
    const bf16* brow0 = Bt + (size_t)(n0 + r) * K;
    const bf16* brow1 = Bt + (size_t)(n0 + 16 + r) * K;
    const bf16* brow2 = Bt + (size_t)(n0 + 32 + r) * K;
    const bf16* brow3 = Bt + (size_t)(n0 + 48 + r) * K;

    v8f acc00 = {0.f,0.f,0.f,0.f,0.f,0.f,0.f,0.f};
    v8f acc01 = acc00, acc02 = acc00, acc03 = acc00;
    v8f acc10 = acc00, acc11 = acc00, acc12 = acc00, acc13 = acc00;

    for (int k0 = 0; k0 < K; k0 += 32) {
        v16bf a0 = load_a_frag(arow0, k0, hf);
        v16bf a1 = load_a_frag(arow1, k0, hf);
        v16bf b0 = load_b_frag(brow0, k0, hf);
        v16bf b1 = load_b_frag(brow1, k0, hf);
        v16bf b2 = load_b_frag(brow2, k0, hf);
        v16bf b3 = load_b_frag(brow3, k0, hf);

        acc00 = __builtin_amdgcn_wmma_f32_16x16x32_bf16(false, a0, false, b0, (short)0, acc00, false, false);
        acc01 = __builtin_amdgcn_wmma_f32_16x16x32_bf16(false, a0, false, b1, (short)0, acc01, false, false);
        acc02 = __builtin_amdgcn_wmma_f32_16x16x32_bf16(false, a0, false, b2, (short)0, acc02, false, false);
        acc03 = __builtin_amdgcn_wmma_f32_16x16x32_bf16(false, a0, false, b3, (short)0, acc03, false, false);
        acc10 = __builtin_amdgcn_wmma_f32_16x16x32_bf16(false, a1, false, b0, (short)0, acc10, false, false);
        acc11 = __builtin_amdgcn_wmma_f32_16x16x32_bf16(false, a1, false, b1, (short)0, acc11, false, false);
        acc12 = __builtin_amdgcn_wmma_f32_16x16x32_bf16(false, a1, false, b2, (short)0, acc12, false, false);
        acc13 = __builtin_amdgcn_wmma_f32_16x16x32_bf16(false, a1, false, b3, (short)0, acc13, false, false);
    }

    // C vgpr i <-> row (m-base + i + hf*8), col (n-base + r)
#pragma unroll
    for (int i = 0; i < 8; ++i) {
        const size_t r0 = (size_t)(m0 + i + hf * 8);
        const size_t r1 = (size_t)(m0 + 16 + i + hf * 8);
        C[r0 * N + n0 + r]      = acc00[i];
        C[r0 * N + n0 + 16 + r] = acc01[i];
        C[r0 * N + n0 + 32 + r] = acc02[i];
        C[r0 * N + n0 + 48 + r] = acc03[i];
        C[r1 * N + n0 + r]      = acc10[i];
        C[r1 * N + n0 + 16 + r] = acc11[i];
        C[r1 * N + n0 + 32 + r] = acc12[i];
        C[r1 * N + n0 + 48 + r] = acc13[i];
    }
}

// ---------------------------------------------------------------------------
// Elementwise helpers (grid-stride).
// ---------------------------------------------------------------------------
__global__ void cast_bf16_kernel(const float* __restrict__ in, bf16* __restrict__ out, int n) {
    for (int i = blockIdx.x * blockDim.x + threadIdx.x; i < n; i += gridDim.x * blockDim.x)
        out[i] = (bf16)in[i];
}

__global__ void add_inplace_kernel(float* __restrict__ x, const float* __restrict__ y, int n) {
    for (int i = blockIdx.x * blockDim.x + threadIdx.x; i < n; i += gridDim.x * blockDim.x)
        x[i] += y[i];
}

__global__ void silu_mul_kernel(const float* __restrict__ g, const float* __restrict__ u,
                                bf16* __restrict__ o, int n) {
    for (int i = blockIdx.x * blockDim.x + threadIdx.x; i < n; i += gridDim.x * blockDim.x) {
        const float gg = g[i];
        const float s = gg / (1.0f + __expf(-gg));
        o[i] = (bf16)(s * u[i]);
    }
}

// ---------------------------------------------------------------------------
// Host orchestration
// ---------------------------------------------------------------------------
extern "C" void kernel_launch(void* const* d_in, const int* in_sizes, int n_in,
                              void* d_out, int out_size, void* d_ws, size_t ws_size,
                              hipStream_t stream) {
    (void)in_sizes; (void)n_in; (void)out_size; (void)ws_size;

    const int*   ids  = (const int*)d_in[0];
    const float* emb  = (const float*)d_in[1];
    const float* Wq   = (const float*)d_in[2];
    const float* Wk   = (const float*)d_in[3];
    const float* Wv   = (const float*)d_in[4];
    const float* Wo   = (const float*)d_in[5];
    const float* qn   = (const float*)d_in[6];
    const float* kn   = (const float*)d_in[7];
    const float* n1   = (const float*)d_in[8];
    const float* n2   = (const float*)d_in[9];
    const float* Wg   = (const float*)d_in[10];
    const float* Wu   = (const float*)d_in[11];
    const float* Wd   = (const float*)d_in[12];
    const float* fn   = (const float*)d_in[13];
    const float* Wout = (const float*)d_in[14];
    float* out = (float*)d_out;

    // ---- workspace bump allocator (256B aligned) ----
    char* p = (char*)d_ws;
    auto alloc = [&](size_t bytes) -> char* {
        char* r = p;
        p += (bytes + 255) & ~(size_t)255;
        return r;
    };
    // bf16 transposed weights (N x K per matrix)
    bf16* wtq   = (bf16*)alloc((size_t)Ln * Dm * Dm * 2);
    bf16* wtk   = (bf16*)alloc((size_t)Ln * (KVn * HDn) * Dm * 2);
    bf16* wtv   = (bf16*)alloc((size_t)Ln * (KVn * HDn) * Dm * 2);
    bf16* wto   = (bf16*)alloc((size_t)Ln * Dm * Dm * 2);
    bf16* wtg   = (bf16*)alloc((size_t)Ln * FFn * Dm * 2);
    bf16* wtu   = (bf16*)alloc((size_t)Ln * FFn * Dm * 2);
    bf16* wtd   = (bf16*)alloc((size_t)Ln * Dm * FFn * 2);
    bf16* wtout = (bf16*)alloc((size_t)Vn * Dm * 2);
    // activations
    float* x    = (float*)alloc((size_t)Tn * Dm * 4);
    bf16*  hb   = (bf16*) alloc((size_t)Tn * Dm * 2);
    float* qf   = (float*)alloc((size_t)Tn * Hn * HDn * 4);
    float* kf   = (float*)alloc((size_t)Tn * KVn * HDn * 4);
    float* vf   = (float*)alloc((size_t)Tn * KVn * HDn * 4);
    float* qr   = (float*)alloc((size_t)Tn * Hn * HDn * 4);
    float* kr   = (float*)alloc((size_t)Tn * KVn * HDn * 4);
    float* ctxf = (float*)alloc((size_t)Tn * Dm * 4);
    bf16*  ctxb = (bf16*) alloc((size_t)Tn * Dm * 2);
    float* tmp  = (float*)alloc((size_t)Tn * Dm * 4);
    float* gf   = (float*)alloc((size_t)Tn * FFn * 4);
    float* uf   = (float*)alloc((size_t)Tn * FFn * 4);
    bf16*  gub  = (bf16*) alloc((size_t)Tn * FFn * 2);

    const dim3 tb(32, 8);
    auto tcvt = [&](const float* W, bf16* Wt, int K, int N) {
        transpose_cvt_kernel<<<dim3(N / 32, K / 32), tb, 0, stream>>>(W, Wt, K, N);
    };
    auto gemm = [&](const bf16* A, const bf16* Bt, float* C, int M, int N, int K) {
        gemm_bf16_wmma<<<dim3(N / 256, M / 64), 256, 0, stream>>>(A, Bt, C, M, N, K);
    };

    // ---- one-time (per launch) weight convert + transpose ----
    for (int l = 0; l < Ln; ++l) {
        tcvt(Wq + (size_t)l * Dm * Dm,            wtq + (size_t)l * Dm * Dm,            Dm, Dm);
        tcvt(Wk + (size_t)l * Dm * (KVn * HDn),   wtk + (size_t)l * (KVn * HDn) * Dm,   Dm, KVn * HDn);
        tcvt(Wv + (size_t)l * Dm * (KVn * HDn),   wtv + (size_t)l * (KVn * HDn) * Dm,   Dm, KVn * HDn);
        tcvt(Wo + (size_t)l * Dm * Dm,            wto + (size_t)l * Dm * Dm,            Dm, Dm);
        tcvt(Wg + (size_t)l * Dm * FFn,           wtg + (size_t)l * FFn * Dm,           Dm, FFn);
        tcvt(Wu + (size_t)l * Dm * FFn,           wtu + (size_t)l * FFn * Dm,           Dm, FFn);
        tcvt(Wd + (size_t)l * FFn * Dm,           wtd + (size_t)l * Dm * FFn,           FFn, Dm);
    }
    tcvt(Wout, wtout, Dm, Vn);

    // ---- embedding ----
    embed_kernel<<<Tn, 256, 0, stream>>>(ids, emb, x);

    // ---- transformer layers ----
    for (int l = 0; l < Ln; ++l) {
        // attention sub-block
        rmsnorm_bf16_kernel<<<Tn, 256, 0, stream>>>(x, n1 + (size_t)l * Dm, hb);
        gemm(hb, wtq + (size_t)l * Dm * Dm,          qf, Tn, Hn * HDn,  Dm);
        gemm(hb, wtk + (size_t)l * (KVn * HDn) * Dm, kf, Tn, KVn * HDn, Dm);
        gemm(hb, wtv + (size_t)l * (KVn * HDn) * Dm, vf, Tn, KVn * HDn, Dm);
        rmsnorm_rope_kernel<<<Tn * Hn,  64, 0, stream>>>(qf, qn + (size_t)l * HDn, qr, Hn);
        rmsnorm_rope_kernel<<<Tn * KVn, 64, 0, stream>>>(kf, kn + (size_t)l * HDn, kr, KVn);
        attention_kernel<<<dim3(Sn, Hn, Bn), 128, 0, stream>>>(qr, kr, vf, ctxf);
        cast_bf16_kernel<<<4096, 256, 0, stream>>>(ctxf, ctxb, Tn * Dm);
        gemm(ctxb, wto + (size_t)l * Dm * Dm, tmp, Tn, Dm, Dm);
        add_inplace_kernel<<<4096, 256, 0, stream>>>(x, tmp, Tn * Dm);
        // MLP sub-block
        rmsnorm_bf16_kernel<<<Tn, 256, 0, stream>>>(x, n2 + (size_t)l * Dm, hb);
        gemm(hb, wtg + (size_t)l * FFn * Dm, gf, Tn, FFn, Dm);
        gemm(hb, wtu + (size_t)l * FFn * Dm, uf, Tn, FFn, Dm);
        silu_mul_kernel<<<8192, 256, 0, stream>>>(gf, uf, gub, Tn * FFn);
        gemm(gub, wtd + (size_t)l * Dm * FFn, tmp, Tn, Dm, FFn);
        add_inplace_kernel<<<4096, 256, 0, stream>>>(x, tmp, Tn * Dm);
    }

    // ---- final norm + LM head ----
    rmsnorm_bf16_kernel<<<Tn, 256, 0, stream>>>(x, fn, hb);
    gemm(hb, wtout, out, Tn, Vn, Dm);
}
